// LayerNormMLP_65077344469262
// MI455X (gfx1250) — matrix operand order repaired and checked by reading, not compiled, not run on hardware
//
#include <hip/hip_runtime.h>
#include <hip/hip_bf16.h>

// ---------------------------------------------------------------------------
// LayerNorm + MLP (4096 x 4096 -> 16384 -> 4096), bf16 WMMA pipeline with
// async global->LDS staging (GLOBAL_LOAD_ASYNC_TO_LDS_B128, ASYNCcnt).
// ---------------------------------------------------------------------------

typedef __bf16 v16bf __attribute__((ext_vector_type(16)));
typedef __bf16 v8bf  __attribute__((ext_vector_type(8)));
typedef float  v8f   __attribute__((ext_vector_type(8)));

#define TOKENS 4096
#define DMODEL 4096
#define FC1DIM 16384
#define FC2DIM 4096

// ---------------- async global->LDS copy primitives (CDNA5) ----------------
#if defined(__HIP_DEVICE_COMPILE__) && defined(__gfx1250__)
#define CDNA5_ASYNC_LDS 1
#endif

typedef __attribute__((address_space(3))) void lds_void_t;

__device__ __forceinline__ void async_copy16(void* ldsp, const void* gp) {
#ifdef CDNA5_ASYNC_LDS
  // LDS operand wants the 32-bit LDS byte address (AS(3) offset).
  unsigned loff = (unsigned)(unsigned long long)(lds_void_t*)ldsp;
  asm volatile("global_load_async_to_lds_b128 %0, %1, off"
               :: "v"(loff), "v"((unsigned long long)gp)
               : "memory");
#else
  *(uint4*)ldsp = *(const uint4*)gp;
#endif
}

template <int N>
__device__ __forceinline__ void wait_asynccnt() {
#ifdef CDNA5_ASYNC_LDS
  asm volatile("s_wait_asynccnt %0" :: "i"(N) : "memory");
#endif
}

// ------------------------------ LayerNorm ----------------------------------
__global__ __launch_bounds__(256)
void ln_bf16_kernel(const float* __restrict__ x,
                    const float* __restrict__ gamma,
                    const float* __restrict__ beta,
                    __bf16* __restrict__ y) {
  __shared__ float red_s[256];
  __shared__ float red_q[256];
  const int row = blockIdx.x;
  const int tid = threadIdx.x;
  const float* xr = x + (size_t)row * DMODEL;

  float4 v[4];
  float s = 0.f, q = 0.f;
  const float4* xr4 = (const float4*)xr;
#pragma unroll
  for (int i = 0; i < 4; ++i) {
    v[i] = xr4[tid * 4 + i];
    s += v[i].x + v[i].y + v[i].z + v[i].w;
    q += v[i].x * v[i].x + v[i].y * v[i].y + v[i].z * v[i].z + v[i].w * v[i].w;
  }
  red_s[tid] = s;
  red_q[tid] = q;
  __syncthreads();
#pragma unroll
  for (int off = 128; off > 0; off >>= 1) {
    if (tid < off) {
      red_s[tid] += red_s[tid + off];
      red_q[tid] += red_q[tid + off];
    }
    __syncthreads();
  }
  const float inv_d = 1.0f / (float)DMODEL;
  const float mu = red_s[0] * inv_d;
  const float var = red_q[0] * inv_d - mu * mu;
  const float rs = rsqrtf(var + 1e-3f);

  __bf16 ov[16];
  const float4* g4 = (const float4*)gamma;
  const float4* b4 = (const float4*)beta;
#pragma unroll
  for (int i = 0; i < 4; ++i) {
    float4 g = g4[tid * 4 + i];
    float4 b = b4[tid * 4 + i];
    ov[i * 4 + 0] = (__bf16)((v[i].x - mu) * rs * g.x + b.x);
    ov[i * 4 + 1] = (__bf16)((v[i].y - mu) * rs * g.y + b.y);
    ov[i * 4 + 2] = (__bf16)((v[i].z - mu) * rs * g.z + b.z);
    ov[i * 4 + 3] = (__bf16)((v[i].w - mu) * rs * g.w + b.w);
  }
  uint4* dst = (uint4*)(y + (size_t)row * DMODEL + tid * 16);
  dst[0] = ((uint4*)ov)[0];
  dst[1] = ((uint4*)ov)[1];
}

// ------------------- fp32 -> bf16 transposing converter --------------------
__global__ __launch_bounds__(256)
void transpose_cvt_kernel(const float* __restrict__ in,
                          __bf16* __restrict__ out,
                          int R, int C) {
  __shared__ float tile[32][33];
  const int c0 = blockIdx.x * 32;
  const int r0 = blockIdx.y * 32;
  const int tx = threadIdx.x;
  const int ty = threadIdx.y;
#pragma unroll
  for (int i = 0; i < 32; i += 8)
    tile[ty + i][tx] = in[(size_t)(r0 + ty + i) * C + (c0 + tx)];
  __syncthreads();
#pragma unroll
  for (int i = 0; i < 32; i += 8)
    out[(size_t)(c0 + ty + i) * R + (r0 + tx)] = (__bf16)tile[tx][ty + i];
}

// ------------------------------- GELU --------------------------------------
__device__ __forceinline__ float gelu_tanh(float x) {
  const float c = 0.7978845608028654f;  // sqrt(2/pi)
  float t = tanhf(c * (x + 0.044715f * x * x * x));
  return 0.5f * x * (1.0f + t);
}

// ------------------------------- GEMM --------------------------------------
// C[M,N] = A[M,K] * B[N,K]^T  (+bias, optional GELU)
// Block tile 128x128, K-slab 64, double-buffered LDS filled by async DMA.
// 256 threads = 8 waves (2 in M x 4 in N); each wave: 4x2 tiles of 16x16.
#define BM 128
#define BN 128
#define BK 64
#define LDT 72   // padded LDS row length (144 B, 16 B aligned)

template <bool GELU, typename OutT>
__global__ __launch_bounds__(256)
void gemm_bf16_kernel(const __bf16* __restrict__ A,
                      const __bf16* __restrict__ Bt,
                      const float* __restrict__ bias,
                      OutT* __restrict__ C,
                      int M, int N, int K) {
  __shared__ __align__(16) __bf16 As[2][BM * LDT];
  __shared__ __align__(16) __bf16 Bs[2][BN * LDT];

  const int tid  = threadIdx.x;
  const int lane = tid & 31;
  const int wid  = tid >> 5;
  const int wm   = (wid & 1) * 64;
  const int wn   = (wid >> 1) * 32;
  const int m0   = blockIdx.y * BM;
  const int n0   = blockIdx.x * BN;

  const int row16 = lane & 15;
  const int half  = lane >> 4;

  // This thread's 4 staging chunks (row, k-chunk) are fixed across slabs.
  int srow[4], skc[4];
#pragma unroll
  for (int i = 0; i < 4; ++i) {
    const int c = tid + 256 * i;
    srow[i] = c >> 3;
    skc[i]  = (c & 7) * 8;
  }

  // Issue 8 async b128 DMAs (4 for A, 4 for B) for one K-slab into buf.
  auto stage = [&](int buf, int kb) {
#pragma unroll
    for (int i = 0; i < 4; ++i) {
      async_copy16(&As[buf][srow[i] * LDT + skc[i]],
                   &A[(size_t)(m0 + srow[i]) * K + kb + skc[i]]);
      async_copy16(&Bs[buf][srow[i] * LDT + skc[i]],
                   &Bt[(size_t)(n0 + srow[i]) * K + kb + skc[i]]);
    }
  };

  v8f acc[4][2] = {};
  const int nk = K / BK;

  stage(0, 0);

  for (int t = 0; t < nk; ++t) {
    const int cur = t & 1;
    if (t + 1 < nk) {
      stage(cur ^ 1, (t + 1) * BK);
      wait_asynccnt<8>();   // previous slab's 8 DMAs complete (in-order)
    } else {
      wait_asynccnt<0>();
    }
    __syncthreads();        // all waves' DMA data visible in LDS

#pragma unroll
    for (int kk = 0; kk < BK; kk += 32) {
      v16bf af[4], bf[2];
#pragma unroll
      for (int tm = 0; tm < 4; ++tm) {
        // A 16x32 fragment (ISA 7.12.2): lane(row16,half) holds K [half*8,+8)
        // and [16+half*8,+8)
        const __bf16* base = &As[cur][(wm + tm * 16 + row16) * LDT + kk + half * 8];
        ((v8bf*)&af[tm])[0] = *(const v8bf*)(base);
        ((v8bf*)&af[tm])[1] = *(const v8bf*)(base + 16);
      }
#pragma unroll
      for (int tn = 0; tn < 2; ++tn) {
        // B 32x16 fragment: lane(col=row16,half) holds K [half*8,+8) and [16+half*8,+8)
        const __bf16* base = &Bs[cur][(wn + tn * 16 + row16) * LDT + kk + half * 8];
        ((v8bf*)&bf[tn])[0] = *(const v8bf*)(base);
        ((v8bf*)&bf[tn])[1] = *(const v8bf*)(base + 16);
      }
#pragma unroll
      for (int tm = 0; tm < 4; ++tm)
#pragma unroll
        for (int tn = 0; tn < 2; ++tn)
          acc[tm][tn] = __builtin_amdgcn_wmma_f32_16x16x32_bf16(
              false, af[tm], false, bf[tn], (short)0, acc[tm][tn], false, false);
    }
    __syncthreads();        // everyone done reading buf before it is refilled
  }

  // ---- epilogue: bias (+ GELU), store ----
  const int colc   = lane & 15;
  const int rowadd = (lane >> 4) * 8;   // C/D layout: lanes 16-31 hold M=r+8
#pragma unroll
  for (int tm = 0; tm < 4; ++tm) {
#pragma unroll
    for (int tn = 0; tn < 2; ++tn) {
      const int col = n0 + wn + tn * 16 + colc;
      const float bv = bias[col];
#pragma unroll
      for (int r = 0; r < 8; ++r) {
        const int row = m0 + wm + tm * 16 + rowadd + r;
        float v = acc[tm][tn][r] + bv;
        if (GELU) v = gelu_tanh(v);
        C[(size_t)row * N + col] = (OutT)v;
      }
    }
  }
}

// ------------------------------ launcher -----------------------------------
extern "C" void kernel_launch(void* const* d_in, const int* in_sizes, int n_in,
                              void* d_out, int out_size, void* d_ws, size_t ws_size,
                              hipStream_t stream) {
  const float* x     = (const float*)d_in[0];  // [T, D]
  const float* gamma = (const float*)d_in[1];  // [D]
  const float* beta  = (const float*)d_in[2];  // [D]
  const float* w1    = (const float*)d_in[3];  // [D, FC1]
  const float* b1    = (const float*)d_in[4];  // [FC1]
  const float* w2    = (const float*)d_in[5];  // [FC1, FC2]
  const float* b2    = (const float*)d_in[6];  // [FC2]
  float* out = (float*)d_out;                  // [T, FC2]

  char* ws = (char*)d_ws;
  __bf16* ln  = (__bf16*)(ws);                              // 32 MB  [T][D]
  __bf16* w1t = (__bf16*)(ws + ((size_t)32 << 20));         // 128 MB [FC1][D]
  __bf16* act = (__bf16*)(ws + ((size_t)160 << 20));        // 128 MB [T][FC1]
  __bf16* w2t = (__bf16*)(ws + ((size_t)288 << 20));        // 128 MB [FC2][FC1]

  // 1) LayerNorm -> bf16
  ln_bf16_kernel<<<TOKENS, 256, 0, stream>>>(x, gamma, beta, ln);

  // 2) weights fp32 -> bf16, transposed to K-major [N][K]
  transpose_cvt_kernel<<<dim3(FC1DIM / 32, DMODEL / 32), dim3(32, 8), 0, stream>>>(
      w1, w1t, DMODEL, FC1DIM);
  transpose_cvt_kernel<<<dim3(FC2DIM / 32, FC1DIM / 32), dim3(32, 8), 0, stream>>>(
      w2, w2t, FC1DIM, FC2DIM);

  // 3) GEMM1 + bias + GELU -> bf16 activations
  gemm_bf16_kernel<true, __bf16>
      <<<dim3(FC1DIM / BN, TOKENS / BM), 256, 0, stream>>>(
          ln, w1t, b1, act, TOKENS, FC1DIM, DMODEL);

  // 4) GEMM2 + bias -> fp32 output
  gemm_bf16_kernel<false, float>
      <<<dim3(FC2DIM / BN, TOKENS / BM), 256, 0, stream>>>(
          act, w2t, b2, out, TOKENS, FC2DIM, FC1DIM);
}
